// MLP_or_nextstep_6167573037697
// MI455X (gfx1250) — matrix operand order repaired and checked by reading, not compiled, not run on hardware
//
#include <hip/hip_runtime.h>

typedef __attribute__((ext_vector_type(16))) _Float16 v16h;
typedef __attribute__((ext_vector_type(8)))  float    v8f;

#define TT    1024
#define NROW  1019            // T - WS output rows

static __device__ __forceinline__ float fast_tanh(float x) {
#if __has_builtin(__builtin_amdgcn_tanhf)
    return __builtin_amdgcn_tanhf(x);
#elif __has_builtin(__builtin_amdgcn_exp2f)
    float e = __builtin_amdgcn_exp2f(2.885390081777927f * x);   // exp(2x)
    return 1.0f - 2.0f / (e + 1.0f);
#else
    float e = exp2f(2.885390081777927f * x);
    return 1.0f - 2.0f / (e + 1.0f);
#endif
}

// Pin a packed operand in VGPRs exactly once (volatile: cannot be sunk into loops,
// cannot be rematerialized -> the 5 phase matrices stay resident, no per-step rebuild).
static __device__ __forceinline__ void pin_once(v16h& v) { asm volatile("" : "+v"(v)); }

// Shared (read-only) weight state, one copy per wave:
struct Weights {
    v16h  a1[5];              // phase-rotated W1 A-matrices (row j in lane j / 16+j)
    float w2a[6], w2b[6];     // W2 rows (uniform -> SGPRs)
    float b2a, b2b;
};

// Per-chain rollout state. B1 layout (Kn = element n, same slots in all 32 lanes):
//   K0..4  = u window slots   K5 = 1.0 (layer-1 bias column)
//   K6..10 = p0 window slots  K11..15 = p1 window slots
// A1 weights only el0..7 in lanes 0-15 and el8..15 in lanes 16-31; the rest are zero-weighted.
struct Chain {
    v16h  B1;
    const float* base;        // traj row base for this chain's batch tile
    float* op;                // per-lane output cursor (lanes 0-15 store)
};

// One fused rollout step (both chains) at compile-time phase PHI (= step mod 5).
// Explicit issue order: WMMAs together -> independent u-inserts (hazard fillers)
// -> all tanhs -> 4-way-ILP FMA reduction -> pred inserts.
template<int PHI>
static __device__ __forceinline__ void do_step2(const Weights& W, Chain& cA, Chain& cB,
                                                float uA, float uB,
                                                float2& rA, float2& rB) {
    v8f c0 = {};
    v8f dA = __builtin_amdgcn_wmma_f32_16x16x32_f16(
        false, W.a1[PHI], false, cA.B1, (short)0, c0, false, false);
    v8f dB = __builtin_amdgcn_wmma_f32_16x16x32_f16(
        false, W.a1[PHI], false, cB.B1, (short)0, c0, false, false);

    // WAR-safe u-slot overwrites: independent of this step's results, fill hazard slots
    cA.B1[PHI] = (_Float16)uA;
    cB.B1[PHI] = (_Float16)uB;

    float a0 = fast_tanh(dA[0]);
    float a1 = fast_tanh(dA[1]);
    float a2 = fast_tanh(dA[2]);
    float a3 = fast_tanh(dA[3]);
    float a4 = fast_tanh(dA[4]);
    float a5 = fast_tanh(dA[5]);
    float b0 = fast_tanh(dB[0]);
    float b1 = fast_tanh(dB[1]);
    float b2 = fast_tanh(dB[2]);
    float b3 = fast_tanh(dB[3]);
    float b4 = fast_tanh(dB[4]);
    float b5 = fast_tanh(dB[5]);

    // layer 2: 4 independent accumulator chains, j-major for ILP
    float pA0 = W.b2a, pA1 = W.b2b, pB0 = W.b2a, pB1 = W.b2b;
    pA0 = fmaf(a0, W.w2a[0], pA0); pA1 = fmaf(a0, W.w2b[0], pA1);
    pB0 = fmaf(b0, W.w2a[0], pB0); pB1 = fmaf(b0, W.w2b[0], pB1);
    pA0 = fmaf(a1, W.w2a[1], pA0); pA1 = fmaf(a1, W.w2b[1], pA1);
    pB0 = fmaf(b1, W.w2a[1], pB0); pB1 = fmaf(b1, W.w2b[1], pB1);
    pA0 = fmaf(a2, W.w2a[2], pA0); pA1 = fmaf(a2, W.w2b[2], pA1);
    pB0 = fmaf(b2, W.w2a[2], pB0); pB1 = fmaf(b2, W.w2b[2], pB1);
    pA0 = fmaf(a3, W.w2a[3], pA0); pA1 = fmaf(a3, W.w2b[3], pA1);
    pB0 = fmaf(b3, W.w2a[3], pB0); pB1 = fmaf(b3, W.w2b[3], pB1);
    pA0 = fmaf(a4, W.w2a[4], pA0); pA1 = fmaf(a4, W.w2b[4], pA1);
    pB0 = fmaf(b4, W.w2a[4], pB0); pB1 = fmaf(b4, W.w2b[4], pB1);
    pA0 = fmaf(a5, W.w2a[5], pA0); pA1 = fmaf(a5, W.w2b[5], pA1);
    pB0 = fmaf(b5, W.w2a[5], pB0); pB1 = fmaf(b5, W.w2b[5], pB1);

    rA.x = pA0; rA.y = pA1;
    rB.x = pB0; rB.y = pB1;

    // circular-slot pred overwrites, fully lane-local
    cA.B1[6 + PHI]  = (_Float16)pA0;
    cA.B1[11 + PHI] = (_Float16)pA1;
    cB.B1[6 + PHI]  = (_Float16)pB0;
    cB.B1[11 + PHI] = (_Float16)pB1;
}

// Warm-up helper: overwrite u-slot I (zero-weighted in hi lanes -> unmasked).
template<int I>
static __device__ __forceinline__ void fixu(Chain& c, float uval) {
    c.B1[I] = (_Float16)uval;
}

// Seed windows: u0..4 -> K0..4, v0..4 -> K6..10, w0..4 -> K11..15, 1.0 -> K5.
static __device__ __forceinline__ void seed(Chain& c) {
    v16h b = {};
#pragma unroll
    for (int i = 0; i < 5; ++i) {
        b[i]      = (_Float16)c.base[i * 3 + 0];
        b[6 + i]  = (_Float16)c.base[i * 3 + 1];
        b[11 + i] = (_Float16)c.base[i * 3 + 2];
    }
    b[5] = (_Float16)1.0f;
    c.B1 = b;
}

// Fused warm-up, steps s=0..4 with the reference's irregular u indexing.
static __device__ __forceinline__ void warmup2(const Weights& W, Chain& cA, Chain& cB,
                                               bool lo) {
    float uA[12], uB[12];
#pragma unroll
    for (int i = 0; i < 12; ++i) { uA[i] = cA.base[i * 3]; uB[i] = cB.base[i * 3]; }

    float2 wA[5], wB[5];
    do_step2<0>(W, cA, cB, uA[5],  uB[5],  wA[0], wB[0]);                 // s=0
    do_step2<1>(W, cA, cB, uA[7],  uB[7],  wA[1], wB[1]);                 // s=1
        fixu<0>(cA, uA[6]);  fixu<0>(cB, uB[6]);                          // -> {u2,u3,u4,u6,u7}
    do_step2<2>(W, cA, cB, uA[9],  uB[9],  wA[2], wB[2]);                 // s=2
        fixu<0>(cA, uA[7]);  fixu<1>(cA, uA[8]);
        fixu<0>(cB, uB[7]);  fixu<1>(cB, uB[8]);
    do_step2<3>(W, cA, cB, uA[11], uB[11], wA[3], wB[3]);                 // s=3
        fixu<0>(cA, uA[8]);  fixu<1>(cA, uA[9]);  fixu<2>(cA, uA[10]);
        fixu<0>(cB, uB[8]);  fixu<1>(cB, uB[9]);  fixu<2>(cB, uB[10]);
    do_step2<4>(W, cA, cB, uA[9],  uB[9],  wA[4], wB[4]);                 // s=4
        fixu<0>(cA, uA[5]);  fixu<1>(cA, uA[6]);  fixu<2>(cA, uA[7]);  fixu<3>(cA, uA[8]);
        fixu<0>(cB, uB[5]);  fixu<1>(cB, uB[6]);  fixu<2>(cB, uB[7]);  fixu<3>(cB, uB[8]);

    if (lo) {
        float2* oA = (float2*)cA.op;
        float2* oB = (float2*)cB.op;
        oA[0] = wA[0]; oA[1] = wA[1]; oA[2] = wA[2]; oA[3] = wA[3]; oA[4] = wA[4];
        oB[0] = wB[0]; oB[1] = wB[1]; oB[2] = wB[2]; oB[3] = wB[3]; oB[4] = wB[4];
    }
    cA.op += 10;
    cB.op += 10;
}

// Single-wave workgroups; min-waves hint 1 frees the full VGPR file per wave.
// Two independent rollout chains per wave, manually fused per step so chain B's
// WMMA/tanh work fills chain A's WMMA->VALU hazard slots and vice versa.
__global__ __launch_bounds__(32, 1)
void mlp_rollout_kernel(const float* __restrict__ traj,
                        const float* __restrict__ W1, const float* __restrict__ b1,
                        const float* __restrict__ W2, const float* __restrict__ b2,
                        float* __restrict__ out) {
    int lane  = threadIdx.x & 31;
    int wave  = blockIdx.x;                    // one wave per workgroup
    int bidxA = wave * 32 + (lane & 15);       // chain A: batches [32w, 32w+16)
    int bidxB = bidxA + 16;                    // chain B: batches [32w+16, 32w+32)
    bool lo   = lane < 16;

    Weights W;
    Chain cA, cB;
    cA.base = traj + (size_t)bidxA * (TT * 3);
    cB.base = traj + (size_t)bidxB * (TT * 3);
    cA.op   = out + (size_t)bidxA * (NROW * 2);
    cB.op   = out + (size_t)bidxB * (NROW * 2);

    // ---- layer-2 weights: wave-uniform scalars ----
#pragma unroll
    for (int k = 0; k < 6; ++k) { W.w2a[k] = W2[k]; W.w2b[k] = W2[6 + k]; }
    W.b2a = b2[0]; W.b2b = b2[1];

    // ---- W1 row for this lane (row j = lane&15; rows >= 6 are zero pad) ----
    int  j  = lane & 15;
    bool jv = j < 6;
    int  jj = jv ? j : 0;
    float w1r[15];
#pragma unroll
    for (int k = 0; k < 15; ++k) { float w = W1[jj * 15 + k]; w1r[k] = jv ? w : 0.0f; }
    float b1w = b1[jj]; float b1j = jv ? b1w : 0.0f;

    // ---- 5 phase-rotated A1 matrices (built once, pinned resident in VGPRs) ----
    // lo lanes  (reg r=0..3 -> K0..7, r=4..7 -> K16..23):
    //   el0..4 = u wgts, el5 = b1, el6,7 = p0 wgts slot0,1 ; el8..15 = 0
    // hi lanes  (reg r=0..3 -> K8..15, r=4..7 -> K24..31):
    //   el0..7 = 0 ; el8..10 = p0 wgts slot2..4, el11..15 = p1 wgts slot0..4
#pragma unroll
    for (int ph = 0; ph < 5; ++ph) {
        int pos[5];
#pragma unroll
        for (int i = 0; i < 5; ++i) pos[i] = (i + 5 - ph) % 5;
        v16h lA = {}, hA = {};
#pragma unroll
        for (int i = 0; i < 5; ++i) lA[i] = (_Float16)w1r[pos[i]];
        lA[5] = (_Float16)b1j;
        lA[6] = (_Float16)w1r[5 + pos[0]];
        lA[7] = (_Float16)w1r[5 + pos[1]];
        hA[8] = (_Float16)w1r[5 + pos[2]];
        hA[9] = (_Float16)w1r[5 + pos[3]];
        hA[10] = (_Float16)w1r[5 + pos[4]];
#pragma unroll
        for (int i = 0; i < 5; ++i) hA[11 + i] = (_Float16)w1r[10 + pos[i]];
        W.a1[ph] = lo ? lA : hA;
        pin_once(W.a1[ph]);       // volatile: executes once, cannot be sunk/remat'd
    }

    // ---- seeds + fused warm-up ----
    seed(cA);
    seed(cB);
    warmup2(W, cA, cB, lo);

    // ---- steady scan: steps s = 5..1014, 5-phase unrolled, chains fused ----
#pragma unroll 1
    for (int s5 = 5; s5 <= 1010; s5 += 5) {
        float unA[5], unB[5];
#pragma unroll
        for (int i = 0; i < 5; ++i) {
            unA[i] = cA.base[(s5 + 5 + i) * 3];
            unB[i] = cB.base[(s5 + 5 + i) * 3];
        }
        float2 wA[5], wB[5];
        do_step2<0>(W, cA, cB, unA[0], unB[0], wA[0], wB[0]);
        do_step2<1>(W, cA, cB, unA[1], unB[1], wA[1], wB[1]);
        do_step2<2>(W, cA, cB, unA[2], unB[2], wA[2], wB[2]);
        do_step2<3>(W, cA, cB, unA[3], unB[3], wA[3], wB[3]);
        do_step2<4>(W, cA, cB, unA[4], unB[4], wA[4], wB[4]);
        if (lo) {
            float2* oA = (float2*)cA.op;
            float2* oB = (float2*)cB.op;
            oA[0] = wA[0]; oA[1] = wA[1]; oA[2] = wA[2]; oA[3] = wA[3]; oA[4] = wA[4];
            oB[0] = wB[0]; oB[1] = wB[1]; oB[2] = wB[2]; oB[3] = wB[3]; oB[4] = wB[4];
        }
        cA.op += 10;
        cB.op += 10;
    }

    // ---- tail steps 1015..1018 (phases 0..3), chains fused ----
    {
        float tA[4], tB[4];
#pragma unroll
        for (int i = 0; i < 4; ++i) {
            tA[i] = cA.base[(1020 + i) * 3];
            tB[i] = cB.base[(1020 + i) * 3];
        }
        float2 wA[4], wB[4];
        do_step2<0>(W, cA, cB, tA[0], tB[0], wA[0], wB[0]);
        do_step2<1>(W, cA, cB, tA[1], tB[1], wA[1], wB[1]);
        do_step2<2>(W, cA, cB, tA[2], tB[2], wA[2], wB[2]);
        do_step2<3>(W, cA, cB, tA[3], tB[3], wA[3], wB[3]);
        if (lo) {
            float2* oA = (float2*)cA.op;
            float2* oB = (float2*)cB.op;
            oA[0] = wA[0]; oA[1] = wA[1]; oA[2] = wA[2]; oA[3] = wA[3];
            oB[0] = wB[0]; oB[1] = wB[1]; oB[2] = wB[2]; oB[3] = wB[3];
        }
    }
}

extern "C" void kernel_launch(void* const* d_in, const int* in_sizes, int n_in,
                              void* d_out, int out_size, void* d_ws, size_t ws_size,
                              hipStream_t stream) {
    const float* traj = (const float*)d_in[0];
    const float* W1   = (const float*)d_in[1];
    const float* b1   = (const float*)d_in[2];
    const float* W2   = (const float*)d_in[3];
    const float* b2   = (const float*)d_in[4];
    float* out = (float*)d_out;

    // 8192 batch rows, 2 x 16-row chains per wave -> 256 waves, one per workgroup
    dim3 grid(256), block(32);
    mlp_rollout_kernel<<<grid, block, 0, stream>>>(traj, W1, b1, W2, b2, out);
}
